// CrossAnalysisModel_60318520705772
// MI455X (gfx1250) — matrix operand order repaired and checked by reading, not compiled, not run on hardware
//
#include <hip/hip_runtime.h>
#include <math.h>
#include <stdint.h>

// ---------------------------------------------------------------------------
// Hetero-GATv2 forward for MI455X (gfx1250, wave32).
//  - All dense linear maps (encoders feed f16 activations; Wl/Wr GEMMs) run on
//    v_wmma_f32_16x16x32_f16: one wave per 16x16 output tile, K-loop step 32.
//  - Edge phase: one wave per edge, coalesced feature access, f32 atomics.
//  - Roofline: ~40 GFLOP of GEMM (WMMA-trivial) vs ~5 GB gathered edge traffic
//    (~250us @ 23.3 TB/s, less with 192MB L2 reuse) => edge phase bound; GEMM
//    precision/f16 conversion is essentially free and keeps the matrix units fed.
// ---------------------------------------------------------------------------

typedef __attribute__((ext_vector_type(16))) _Float16 v16h;
typedef __attribute__((ext_vector_type(8)))  _Float16 v8h;
typedef __attribute__((ext_vector_type(8)))  float    v8f;

#define EPS_BN_F 1e-5f

__device__ __forceinline__ void atomicMaxF(float* addr, float val) {
  int* ia = reinterpret_cast<int*>(addr);
  int old = *ia;
  while (__int_as_float(old) < val) {
    int assumed = old;
    old = atomicCAS(ia, assumed, __float_as_int(val));
    if (old == assumed) break;
  }
}

__global__ void fill_f32_kernel(float* __restrict__ p, long n, float v) {
  long i = (long)blockIdx.x * blockDim.x + threadIdx.x;
  if (i < n) p[i] = v;
}

// W: [K][N] f32 row-major  ->  WT: [N][K] f16 (so B fragments load contiguously)
__global__ void cvt_wT_f16_kernel(const float* __restrict__ W,
                                  _Float16* __restrict__ WT, int K, int N) {
  int i = blockIdx.x * blockDim.x + threadIdx.x;
  if (i >= K * N) return;
  int k = i / N, n = i - k * N;
  WT[(long)n * K + k] = (_Float16)W[i];
}

// h = relu(bn(x @ W + b)) -> f16 activation buffer (64 features)
__global__ void encode_kernel(const float* __restrict__ x, const float* __restrict__ W,
                              const float* __restrict__ b, const float* __restrict__ beta,
                              const float* __restrict__ gamma, const float* __restrict__ mean,
                              const float* __restrict__ var, _Float16* __restrict__ h16,
                              int N, int fin) {
  long i = (long)blockIdx.x * blockDim.x + threadIdx.x;
  if (i >= (long)N * 64) return;
  int n = (int)(i >> 6), j = (int)(i & 63);
  float s = b[j];
  const float* xr = x + (long)n * fin;
  for (int k = 0; k < fin; ++k) s += xr[k] * W[k * 64 + j];
  s = gamma[j] * (s - mean[j]) * rsqrtf(var[j] + EPS_BN_F) + beta[j];
  h16[i] = (_Float16)fmaxf(s, 0.f);
}

// out[M][N] = A[M][K](f16) * B (BT:[N][K] f16) + bias[N], f32 accum via WMMA.
// One wave per 16x16 tile; K in {64,128}, N in {64,128} (multiples of 16/32).
__global__ void gemm_wmma_bias_f16(const _Float16* __restrict__ A,
                                   const _Float16* __restrict__ BT,
                                   const float* __restrict__ bias,
                                   float* __restrict__ out,
                                   int M, int N, int K) {
  int wave = threadIdx.x >> 5;
  int lane = threadIdx.x & 31;
  int tiles_n = N >> 4;
  int tiles_m = (M + 15) >> 4;
  int tile = blockIdx.x * (blockDim.x >> 5) + wave;
  if (tile >= tiles_m * tiles_n) return;
  int tm = (tile / tiles_n) << 4;
  int tn = (tile % tiles_n) << 4;
  int lo = lane & 15, hi = lane >> 4;

  int arow = tm + lo; if (arow >= M) arow = M - 1;   // clamp ragged M (stores guarded)
  int bcol = tn + lo;

  v8f acc = {};
  for (int k0 = 0; k0 < K; k0 += 32) {
    union { v16h v; v8h h[2]; } a, b;
    // A fragment (16-bit A 16x32 layout): lane(lo,hi) row=tm+lo,
    //   elems 0..7  -> K = k0 + 8*hi + 0..7
    //   elems 8..15 -> K = k0 + 16 + 8*hi + 0..7   (two 16B contiguous loads)
    const _Float16* ap = A + (long)arow * K + k0 + 8 * hi;
    a.h[0] = *reinterpret_cast<const v8h*>(ap);
    a.h[1] = *reinterpret_cast<const v8h*>(ap + 16);
    // B fragment (B 32x16): lane(lo,hi) col=tn+lo, same K order as A
    const _Float16* bp = BT + (long)bcol * K + k0 + 8 * hi;
    b.h[0] = *reinterpret_cast<const v8h*>(bp);
    b.h[1] = *reinterpret_cast<const v8h*>(bp + 16);
    acc = __builtin_amdgcn_wmma_f32_16x16x32_f16(false, a.v, false, b.v,
                                                 (short)0, acc, false, false);
  }
  float bj = bias[tn + lo];
#pragma unroll
  for (int r = 0; r < 8; ++r) {
    int m = tm + r + 8 * hi;   // C/D layout: VGPR r -> M = r (lanes 0-15) / 8+r (16-31)
    if (m < M) out[(long)m * N + tn + lo] = acc[r] + bj;
  }
}

// Per edge: logits[e][h] = sum_c leaky(xl[s]+xr[d]+ea*We)[h,c] * att[h,c];
// also segment-max into mx[dst][h]. One wave per edge, lanes cover c.
__global__ void gat_logits_kernel(const float* __restrict__ xl, const float* __restrict__ xr,
                                  const float* __restrict__ ea, const float* __restrict__ We,
                                  const float* __restrict__ att,
                                  const int* __restrict__ src, const int* __restrict__ dst,
                                  float* __restrict__ logits, float* __restrict__ mx,
                                  int E, int H) {
  int e = blockIdx.x * (blockDim.x >> 5) + (threadIdx.x >> 5);
  int lane = threadIdx.x & 31;
  if (e >= E) return;
  int s = src[e], d = dst[e];
  int HC = H << 6;
  float w = ea[e];
  const float* xls = xl + (long)s * HC;
  const float* xrd = xr + (long)d * HC;
  float part0 = 0.f, part1 = 0.f;
  for (int h = 0; h < H; ++h) {
    float acc = 0.f;
#pragma unroll
    for (int cc = 0; cc < 2; ++cc) {          // c = lane, lane+32 (C=64)
      int idx = (h << 6) + lane + (cc << 5);
      float m = xls[idx] + xrd[idx] + w * We[idx];
      m = (m > 0.f) ? m : 0.2f * m;            // leaky_relu 0.2
      acc += m * att[idx];
    }
    if (h == 0) part0 = acc; else part1 = acc;
  }
  for (int off = 16; off > 0; off >>= 1) {
    part0 += __shfl_down(part0, off, 32);
    part1 += __shfl_down(part1, off, 32);
  }
  if (lane == 0) {
    logits[(long)e * H] = part0;
    atomicMaxF(&mx[(long)d * H], part0);
    if (H == 2) {
      logits[(long)e * H + 1] = part1;
      atomicMaxF(&mx[(long)d * H + 1], part1);
    }
  }
}

// ex = exp(logit - mx[dst]); den[dst] += ex
__global__ void gat_softmax_kernel(const float* __restrict__ logits, const float* __restrict__ mx,
                                   const int* __restrict__ dst, float* __restrict__ ex,
                                   float* __restrict__ den, int E, int H) {
  long i = (long)blockIdx.x * blockDim.x + threadIdx.x;
  if (i >= (long)E * H) return;
  int e = (int)(i / H), h = (int)(i - (long)e * H);
  int d = dst[e];
  float v = __expf(logits[i] - mx[(long)d * H + h]);
  ex[i] = v;
  atomicAdd(&den[(long)d * H + h], v);
}

// agg[dst] += alpha * xl[src]; one wave per edge, lanes stride features.
__global__ void gat_aggregate_kernel(const float* __restrict__ ex, const float* __restrict__ den,
                                     const float* __restrict__ xl,
                                     const int* __restrict__ src, const int* __restrict__ dst,
                                     float* __restrict__ agg, int E, int H) {
  int e = blockIdx.x * (blockDim.x >> 5) + (threadIdx.x >> 5);
  int lane = threadIdx.x & 31;
  if (e >= E) return;
  int s = src[e], d = dst[e];
  int HC = H << 6;
  float a0 = ex[(long)e * H] / (den[(long)d * H] + 1e-16f);
  float a1 = (H == 2) ? ex[(long)e * H + 1] / (den[(long)d * H + 1] + 1e-16f) : 0.f;
  const float* xls = xl + (long)s * HC;
  float* ad = agg + (long)d * HC;
  for (int idx = lane; idx < HC; idx += 32) {
    float al = (idx < 64) ? a0 : a1;
    atomicAdd(&ad[idx], al * xls[idx]);
  }
}

// mean over relations + bias sum (+ optional ReLU); emit f32 and/or f16.
__global__ void finalize_kernel(const float* __restrict__ agg,
                                const float* __restrict__ b0, const float* __restrict__ b1,
                                const float* __restrict__ b2, float invR, int relu,
                                float* __restrict__ out32, _Float16* __restrict__ out16,
                                long N, int HC) {
  long i = (long)blockIdx.x * blockDim.x + threadIdx.x;
  if (i >= N * HC) return;
  int j = (int)(i % HC);
  float bs = b0[j];
  if (b1) bs += b1[j];
  if (b2) bs += b2[j];
  float v = (agg[i] + bs) * invR;
  if (relu) v = fmaxf(v, 0.f);
  if (out32) out32[i] = v;
  if (out16) out16[i] = (_Float16)v;
}

static inline unsigned cdivu(long a, long b) { return (unsigned)((a + b - 1) / b); }

extern "C" void kernel_launch(void* const* d_in, const int* in_sizes, int n_in,
                              void* d_out, int out_size, void* d_ws, size_t ws_size,
                              hipStream_t stream) {
  (void)n_in; (void)out_size; (void)ws_size;
  const int NU = in_sizes[0] / 7;
  const int NP = in_sizes[1] / 8;
  const int NC = in_sizes[2] / 5;
  const int Nt[3] = {NU, NP, NC};   // types: 0=user, 1=poi, 2=category
  const float* xin[3] = {(const float*)d_in[0], (const float*)d_in[1], (const float*)d_in[2]};

  // RELS processing order: up, pu, pp, pc, cp
  const int relSorted[5] = {4, 3, 2, 1, 0};  // index in sorted-key param order (cp,pc,pp,pu,up)
  const int relSrcT[5]   = {0, 1, 1, 1, 2};
  const int relDstT[5]   = {1, 0, 1, 2, 1};
  const int relEa[5]     = {3, 4, 5, 6, 7};
  const int relSrcI[5]   = {8, 10, 12, 14, 16};
  const int relDstI[5]   = {9, 11, 13, 15, 17};

  // Params flattened after the 18 arrays, pytree sorted-key order:
  //  c1 (5 rels x {We,Wl,Wr,att,bias,bl,br}), c2 (same), enc (category,poi,user x {W,b,beta,gamma,mean,var})
  const int P0 = 18;
  auto gatp = [&](int layer, int rel, int leaf) -> const float* {
    return (const float*)d_in[P0 + layer * 35 + rel * 7 + leaf];
  };
  auto encp = [&](int sortedNt, int leaf) -> const float* {
    return (const float*)d_in[P0 + 70 + sortedNt * 6 + leaf];
  };

  int Emax = 0;
  for (int r = 0; r < 5; ++r) if (in_sizes[relEa[r]] > Emax) Emax = in_sizes[relEa[r]];

  // ---- carve workspace (256B aligned => 16B-aligned WMMA fragment loads) ----
  uintptr_t wsb = (uintptr_t)d_ws; size_t off = 0;
  auto carve = [&](size_t bytes) -> void* {
    off = (off + 255) & ~(size_t)255;
    void* p = (void*)(wsb + off); off += bytes; return p;
  };
  _Float16* h0[3]; _Float16* h1[3]; float* agg1[3];
  for (int t = 0; t < 3; ++t) h0[t]  = (_Float16*)carve((size_t)Nt[t] * 64  * sizeof(_Float16));
  for (int t = 0; t < 3; ++t) h1[t]  = (_Float16*)carve((size_t)Nt[t] * 128 * sizeof(_Float16));
  for (int t = 0; t < 3; ++t) agg1[t] = (float*)  carve((size_t)Nt[t] * 128 * sizeof(float));
  float* xl  = (float*)carve((size_t)NU * 128 * sizeof(float));
  float* xr  = (float*)carve((size_t)NU * 128 * sizeof(float));
  float* lg  = (float*)carve((size_t)Emax * 2 * sizeof(float));
  float* ex  = (float*)carve((size_t)Emax * 2 * sizeof(float));
  float* mx  = (float*)carve((size_t)NU * 2 * sizeof(float));
  float* den = (float*)carve((size_t)NU * 2 * sizeof(float));
  _Float16* WT[2][5][2];
  for (int l = 0; l < 2; ++l)
    for (int r = 0; r < 5; ++r)
      for (int w = 0; w < 2; ++w)
        WT[l][r][w] = (_Float16*)carve((size_t)64 * 128 * sizeof(_Float16));

  // ---- weight conversion (f32 [K][N] -> f16 [N][K]) ----
  for (int l = 0; l < 2; ++l) {
    int K = (l == 0) ? 64 : 128, N = (l == 0) ? 128 : 64;
    for (int r = 0; r < 5; ++r) {
      cvt_wT_f16_kernel<<<cdivu((long)K * N, 256), 256, 0, stream>>>(gatp(l, r, 1), WT[l][r][0], K, N);
      cvt_wT_f16_kernel<<<cdivu((long)K * N, 256), 256, 0, stream>>>(gatp(l, r, 2), WT[l][r][1], K, N);
    }
  }

  // ---- encoders (Linear + BN(eval) + ReLU) -> f16 ----
  const int encSorted[3] = {2, 1, 0};  // user->'user'(2), poi->'poi'(1), category->'category'(0)
  const int fins[3] = {7, 8, 5};
  for (int t = 0; t < 3; ++t) {
    int ei = encSorted[t];
    encode_kernel<<<cdivu((long)Nt[t] * 64, 256), 256, 0, stream>>>(
        xin[t], encp(ei, 0), encp(ei, 1), encp(ei, 2), encp(ei, 3), encp(ei, 4),
        encp(ei, 5), h0[t], Nt[t], fins[t]);
  }

  float* out_seg[3] = {(float*)d_out,
                       (float*)d_out + (size_t)NU * 64,
                       (float*)d_out + (size_t)NU * 64 + (size_t)NP * 64};

  for (int l = 0; l < 2; ++l) {
    const int H = (l == 0) ? 2 : 1;
    const int HC = H * 64;
    const int Kin = (l == 0) ? 64 : 128;
    _Float16** hin = (l == 0) ? h0 : h1;
    float* agg[3];
    for (int t = 0; t < 3; ++t) agg[t] = (l == 0) ? agg1[t] : out_seg[t];

    for (int t = 0; t < 3; ++t) {
      long n = (long)Nt[t] * HC;
      fill_f32_kernel<<<cdivu(n, 256), 256, 0, stream>>>(agg[t], n, 0.f);
    }

    for (int r = 0; r < 5; ++r) {
      int st = relSrcT[r], dt = relDstT[r];
      int Ns = Nt[st], Nd = Nt[dt];
      int E = in_sizes[relEa[r]];
      int sr = relSorted[r];
      const float* eav  = (const float*)d_in[relEa[r]];
      const int*   srcv = (const int*)d_in[relSrcI[r]];
      const int*   dstv = (const int*)d_in[relDstI[r]];

      long tiles = (long)((Ns + 15) / 16) * (HC / 16);
      gemm_wmma_bias_f16<<<cdivu(tiles, 8), 256, 0, stream>>>(
          hin[st], WT[l][sr][0], gatp(l, sr, 5), xl, Ns, HC, Kin);
      tiles = (long)((Nd + 15) / 16) * (HC / 16);
      gemm_wmma_bias_f16<<<cdivu(tiles, 8), 256, 0, stream>>>(
          hin[dt], WT[l][sr][1], gatp(l, sr, 6), xr, Nd, HC, Kin);

      fill_f32_kernel<<<cdivu((long)Nd * H, 256), 256, 0, stream>>>(mx, (long)Nd * H, -INFINITY);
      fill_f32_kernel<<<cdivu((long)Nd * H, 256), 256, 0, stream>>>(den, (long)Nd * H, 0.f);

      gat_logits_kernel<<<cdivu(E, 8), 256, 0, stream>>>(
          xl, xr, eav, gatp(l, sr, 0), gatp(l, sr, 3), srcv, dstv, lg, mx, E, H);
      gat_softmax_kernel<<<cdivu((long)E * H, 256), 256, 0, stream>>>(
          lg, mx, dstv, ex, den, E, H);
      gat_aggregate_kernel<<<cdivu(E, 8), 256, 0, stream>>>(
          ex, den, xl, srcv, dstv, agg[dt], E, H);
    }

    // dst relations: user <- pu(3); poi <- up(4), pp(2), cp(0); category <- pc(1)
    const float* bu0 = gatp(l, 3, 4);
    const float* bp0 = gatp(l, 4, 4);
    const float* bp1 = gatp(l, 2, 4);
    const float* bp2 = gatp(l, 0, 4);
    const float* bc0 = gatp(l, 1, 4);
    if (l == 0) {
      finalize_kernel<<<cdivu((long)NU * HC, 256), 256, 0, stream>>>(
          agg[0], bu0, nullptr, nullptr, 1.f, 1, nullptr, h1[0], NU, HC);
      finalize_kernel<<<cdivu((long)NP * HC, 256), 256, 0, stream>>>(
          agg[1], bp0, bp1, bp2, 1.f / 3.f, 1, nullptr, h1[1], NP, HC);
      finalize_kernel<<<cdivu((long)NC * HC, 256), 256, 0, stream>>>(
          agg[2], bc0, nullptr, nullptr, 1.f, 1, nullptr, h1[2], NC, HC);
    } else {
      finalize_kernel<<<cdivu((long)NU * HC, 256), 256, 0, stream>>>(
          agg[0], bu0, nullptr, nullptr, 1.f, 0, out_seg[0], nullptr, NU, HC);
      finalize_kernel<<<cdivu((long)NP * HC, 256), 256, 0, stream>>>(
          agg[1], bp0, bp1, bp2, 1.f / 3.f, 0, out_seg[1], nullptr, NP, HC);
      finalize_kernel<<<cdivu((long)NC * HC, 256), 256, 0, stream>>>(
          agg[2], bc0, nullptr, nullptr, 1.f, 0, out_seg[2], nullptr, NC, HC);
    }
  }
}
